// GraphRLAgent_69836168233175
// MI455X (gfx1250) — matrix operand order repaired and checked by reading, not compiled, not run on hardware
//
#include <hip/hip_runtime.h>
#include <hip/hip_bf16.h>

typedef __attribute__((ext_vector_type(16))) _Float16 v16h;
typedef __attribute__((ext_vector_type(8)))  _Float16 v8h;
typedef __attribute__((ext_vector_type(8)))  float    v8f;

#define NEG_SLOPE 0.2f

// ---------------------------------------------------------------------------
// Small utility kernels
// ---------------------------------------------------------------------------
__global__ void fill_f32(float* __restrict__ p, float v, int n) {
    int i = blockIdx.x * blockDim.x + threadIdx.x;
    if (i < n) p[i] = v;
}

__global__ void cvt_f32_to_f16(const float* __restrict__ in, _Float16* __restrict__ out, int n) {
    int i = blockIdx.x * blockDim.x + threadIdx.x;
    if (i < n) out[i] = (_Float16)in[i];
}

// WT[nout*K + k] = W[k*Nout + nout]   (row-major [K,Nout] -> row-major [Nout,K])
__global__ void transpose_cvt_f16(const float* __restrict__ W, _Float16* __restrict__ WT,
                                  int K, int Nout) {
    int i = blockIdx.x * blockDim.x + threadIdx.x;
    if (i >= K * Nout) return;
    int n = i / K;
    int k = i - n * K;
    WT[i] = (_Float16)W[(size_t)k * Nout + n];
}

// ---------------------------------------------------------------------------
// WMMA GEMM:  C[M,Nout] (f32) = A[M,K] (f16, row major) * B (given as BT[Nout,K] f16)
// One wave computes FOUR adjacent 16x16 output tiles (same M tile, 4 N tiles):
// one A fragment feeds 4 WMMAs, quartering redundant A traffic.
// Requires Nout % 64 == 0 (true here: 256, 128).
// ---------------------------------------------------------------------------
__global__ void __launch_bounds__(256)
wmma_gemm_f16(const _Float16* __restrict__ A, const _Float16* __restrict__ BT,
              float* __restrict__ C, int M, int K, int Nout) {
    int gid  = blockIdx.x * blockDim.x + threadIdx.x;
    int wave = gid >> 5;
    int lane = gid & 31;
    int tilesM  = (M + 15) >> 4;
    int groupsN = Nout >> 6;            // 4 tiles of 16 per group
    if (wave >= tilesM * groupsN) return;
    int tM = wave / groupsN;
    int gN = wave - tM * groupsN;

    int half = lane >> 4;   // 0: lanes 0-15, 1: lanes 16-31
    int l15  = lane & 15;

    int rowA = tM * 16 + l15;
    if (rowA >= M) rowA = M - 1;   // clamp (M is a multiple of 16 in practice)

    // A fragment: lane<16 -> elems 0..15 = K k0+0..15
    //             lane>=16 -> elems 0..15 = K k0+16..31
    const _Float16* ap = A + (size_t)rowA * K + half * 16;
    // B fragment for tile t: lane holds column (lane&15) of tile, 16 consecutive K
    const _Float16* bp0 = BT + (size_t)(gN * 64 + l15) * K + half * 16;

    v8f acc0 = {}, acc1 = {}, acc2 = {}, acc3 = {};
    for (int k = 0; k < K; k += 32) {
        v8h alo = *(const v8h*)(ap + k);
        v8h ahi = *(const v8h*)(ap + k + 8);
        v16h av;
#pragma unroll
        for (int i = 0; i < 8; ++i) { av[i] = alo[i]; av[i + 8] = ahi[i]; }
        v16h b0 = *(const v16h*)(bp0 + k);
        v16h b1 = *(const v16h*)(bp0 + (size_t)16 * K + k);
        v16h b2 = *(const v16h*)(bp0 + (size_t)32 * K + k);
        v16h b3 = *(const v16h*)(bp0 + (size_t)48 * K + k);
        acc0 = __builtin_amdgcn_wmma_f32_16x16x32_f16(false, av, false, b0, (short)0, acc0, false, false);
        acc1 = __builtin_amdgcn_wmma_f32_16x16x32_f16(false, av, false, b1, (short)0, acc1, false, false);
        acc2 = __builtin_amdgcn_wmma_f32_16x16x32_f16(false, av, false, b2, (short)0, acc2, false, false);
        acc3 = __builtin_amdgcn_wmma_f32_16x16x32_f16(false, av, false, b3, (short)0, acc3, false, false);
    }

    // C/D layout: VGPR r -> row = half*8 + r, lane -> col = lane&15
    int colBase = gN * 64 + l15;
#pragma unroll
    for (int r = 0; r < 8; ++r) {
        int row = tM * 16 + half * 8 + r;
        if (row < M) {
            float* cr = C + (size_t)row * Nout + colBase;
            cr[0]  = acc0[r];
            cr[16] = acc1[r];
            cr[32] = acc2[r];
            cr[48] = acc3[r];
        }
    }
}

// ---------------------------------------------------------------------------
// Attention score per (node, head): es = <h[n,h,:], a_src[h,:]>, ed likewise
// ---------------------------------------------------------------------------
__global__ void node_scores(const float* __restrict__ h, const float* __restrict__ asr,
                            const float* __restrict__ adt, float* __restrict__ es,
                            float* __restrict__ ed, int N, int H, int C) {
    int i = blockIdx.x * blockDim.x + threadIdx.x;
    if (i >= N * H) return;
    int n  = i / H;
    int hh = i - n * H;
    const float* hp = h + (size_t)n * H * C + (size_t)hh * C;
    const float* as = asr + hh * C;
    const float* ad = adt + hh * C;
    float s1 = 0.f, s2 = 0.f;
    for (int c = 0; c < C; ++c) { float v = hp[c]; s1 += v * as[c]; s2 += v * ad[c]; }
    es[i] = s1;
    ed[i] = s2;
}

__device__ __forceinline__ float leaky(float v) { return v > 0.f ? v : NEG_SLOPE * v; }

__device__ __forceinline__ void atomicMaxF(float* addr, float val) {
    if (val >= 0.f) atomicMax((int*)addr, __float_as_int(val));
    else            atomicMin((unsigned int*)addr, (unsigned int)__float_as_int(val));
}

__device__ __forceinline__ void edge_endpoints(const int* __restrict__ src,
                                               const int* __restrict__ dst,
                                               int e, int E, int& s, int& d) {
    if (e < E) { s = src[e]; d = dst[e]; }
    else       { s = e - E; d = e - E; }   // self loops appended after real edges
}

// Pass 1: segment max of leaky_relu(e_src[src]+e_dst[dst]) into m[dst]
__global__ void edge_max(const int* __restrict__ src, const int* __restrict__ dst,
                         int E, int Etot, const float* __restrict__ es,
                         const float* __restrict__ ed, float* __restrict__ m, int H) {
    int i = blockIdx.x * blockDim.x + threadIdx.x;
    if (i >= Etot * H) return;
    int e  = i / H;
    int hh = i - e * H;
    int s, d; edge_endpoints(src, dst, e, E, s, d);
    float ev = leaky(es[s * H + hh] + ed[d * H + hh]);
    atomicMaxF(&m[d * H + hh], ev);
}

// Pass 2: z[dst] += exp(e - m[dst])
__global__ void edge_expsum(const int* __restrict__ src, const int* __restrict__ dst,
                            int E, int Etot, const float* __restrict__ es,
                            const float* __restrict__ ed, const float* __restrict__ m,
                            float* __restrict__ z, int H) {
    int i = blockIdx.x * blockDim.x + threadIdx.x;
    if (i >= Etot * H) return;
    int e  = i / H;
    int hh = i - e * H;
    int s, d; edge_endpoints(src, dst, e, E, s, d);
    float ev = leaky(es[s * H + hh] + ed[d * H + hh]);
    atomicAdd(&z[d * H + hh], __expf(ev - m[d * H + hh]));
}

// Pass 3: agg[dst,h,:] += h[src,h,:] * alpha
// One block per edge; alpha per head broadcast via LDS; float2-vectorized gather.
__global__ void __launch_bounds__(128)
edge_aggregate(const int* __restrict__ src, const int* __restrict__ dst,
               int E, int Etot, const float* __restrict__ es, const float* __restrict__ ed,
               const float* __restrict__ m, const float* __restrict__ z,
               const float* __restrict__ h, float* __restrict__ agg, int H, int C) {
    int e = blockIdx.x;
    if (e >= Etot) return;
    int s, d; edge_endpoints(src, dst, e, E, s, d);
    int HC = H * C;
    const float* hp = h   + (size_t)s * HC;
    float*       ap = agg + (size_t)d * HC;
    __builtin_prefetch(hp + threadIdx.x * 2, 0, 0);   // global_prefetch_b8 of source row
    __shared__ float alpha[8];
    if (threadIdx.x < (unsigned)H) {
        int hh = threadIdx.x;
        float ev = leaky(es[s * H + hh] + ed[d * H + hh]);
        float w  = __expf(ev - m[d * H + hh]);
        alpha[hh] = w / (z[d * H + hh] + 1e-16f);
    }
    __syncthreads();
    const float2* hp2 = (const float2*)hp;
    for (int j = threadIdx.x; j * 2 < HC; j += blockDim.x) {
        float2 v = hp2[j];
        float  al = alpha[(j * 2) / C];   // C is even: both lanes of the pair share a head
        atomicAdd(&ap[j * 2],     v.x * al);
        atomicAdd(&ap[j * 2 + 1], v.y * al);
    }
}

// out = elu(in + bias); optionally also emit f16 copy (for next-layer GEMM input)
__global__ void bias_elu(const float* __restrict__ in, const float* __restrict__ b,
                         float* __restrict__ of32, _Float16* __restrict__ of16,
                         int n, int F) {
    int i = blockIdx.x * blockDim.x + threadIdx.x;
    if (i >= n) return;
    float v = in[i] + b[i % F];
    v = v > 0.f ? v : expm1f(v);
    if (of32) of32[i] = v;
    if (of16) of16[i] = (_Float16)v;
}

// Mean-pool accumulation: sums[batch[n], c] += h[n, c];  cnt[batch[n]] += 1
__global__ void pool_accum(const float* __restrict__ h, const int* __restrict__ batch,
                           float* __restrict__ sums, float* __restrict__ cnt, int N, int F) {
    int n = blockIdx.x;
    int c = threadIdx.x;
    if (n >= N || c >= F) return;
    int g = batch[n];
    atomicAdd(&sums[(size_t)g * F + c], h[(size_t)n * F + c]);
    if (c == 0) atomicAdd(&cnt[g], 1.0f);
}

// Policy / value heads: out[0 .. G*A) = logits, out[G*A .. G*A+G) = values
__global__ void heads(const float* __restrict__ sums, const float* __restrict__ cnt,
                      const float* __restrict__ Wp, const float* __restrict__ bp,
                      const float* __restrict__ Wv, const float* __restrict__ bv,
                      float* __restrict__ out, int G, int F, int A) {
    int g = blockIdx.x;
    int t = threadIdx.x;
    if (g >= G || t > A) return;
    float cn = fmaxf(cnt[g], 1.0f);
    const float* sg = sums + (size_t)g * F;
    float acc = 0.f;
    if (t < A) {
        for (int k = 0; k < F; ++k) acc += (sg[k] / cn) * Wp[(size_t)k * A + t];
        out[(size_t)g * A + t] = acc + bp[t];
    } else {
        for (int k = 0; k < F; ++k) acc += (sg[k] / cn) * Wv[k];
        out[(size_t)G * A + g] = acc + bv[0];
    }
}

// ---------------------------------------------------------------------------
// Host launcher
// ---------------------------------------------------------------------------
extern "C" void kernel_launch(void* const* d_in, const int* in_sizes, int n_in,
                              void* d_out, int out_size, void* d_ws, size_t ws_size,
                              hipStream_t stream) {
    const float* x      = (const float*)d_in[0];
    const int*   ei     = (const int*)  d_in[1];
    const int*   batch  = (const int*)  d_in[2];
    const float* W1     = (const float*)d_in[3];
    const float* a_src1 = (const float*)d_in[4];
    const float* a_dst1 = (const float*)d_in[5];
    const float* b1     = (const float*)d_in[6];
    const float* W2     = (const float*)d_in[7];
    const float* a_src2 = (const float*)d_in[8];
    const float* a_dst2 = (const float*)d_in[9];
    const float* b2     = (const float*)d_in[10];
    const float* Wp     = (const float*)d_in[11];
    const float* bp     = (const float*)d_in[12];
    const float* Wv     = (const float*)d_in[13];
    const float* bv     = (const float*)d_in[14];

    const int N    = in_sizes[2];              // batch has N entries
    const int F_IN = in_sizes[0] / N;          // 128
    const int E    = in_sizes[1] / 2;          // 800000
    const int HID  = in_sizes[8];              // a_src2 = [1,HID] -> 128
    const int H1   = in_sizes[4] / HID;        // 2
    const int A    = in_sizes[12];             // 32
    const int G    = out_size / (A + 1);       // 64
    const int Etot = E + N;                    // edges + self loops
    const int F1   = H1 * HID;                 // 256

    const int* srcp = ei;
    const int* dstp = ei + E;

    // ---- carve workspace (256B aligned), with reuse across layers ----
    char* wp = (char*)d_ws;
    auto carve = [&](size_t bytes) -> void* {
        void* r = (void*)wp;
        wp += (bytes + 255) & ~(size_t)255;
        return r;
    };
    _Float16* x_h  = (_Float16*)carve((size_t)N * F_IN * 2);
    _Float16* w1t  = (_Float16*)carve((size_t)F1 * F_IN * 2);
    float*    h1   = (float*)   carve((size_t)N * F1 * 4);   // layer1 pre-agg features (reused as h2)
    float*    es1  = (float*)   carve((size_t)N * H1 * 4);
    float*    ed1  = (float*)   carve((size_t)N * H1 * 4);
    float*    m1   = (float*)   carve((size_t)N * H1 * 4);
    float*    z1   = (float*)   carve((size_t)N * H1 * 4);
    float*    agg1 = (float*)   carve((size_t)N * F1 * 4);   // reused: [0,N*HID)=agg2, [N*HID,..)=h2a
    _Float16* h1a  = (_Float16*)carve((size_t)N * F1 * 2);
    _Float16* w2t  = (_Float16*)carve((size_t)HID * F1 * 2);
    float*    es2  = (float*)   carve((size_t)N * 4);
    float*    ed2  = (float*)   carve((size_t)N * 4);
    float*    m2   = (float*)   carve((size_t)N * 4);
    float*    z2   = (float*)   carve((size_t)N * 4);
    float*    pool = (float*)   carve((size_t)G * HID * 4);
    float*    cnt  = (float*)   carve((size_t)G * 4);
    float*    h2   = h1;                        // layer2 pre-agg features
    float*    agg2 = agg1;
    float*    h2a  = agg1 + (size_t)N * HID;    // activated layer2 output (f32)

    auto nblk = [](long n, int b) { return (int)((n + b - 1) / b); };

    // ---- layer 1 GEMM: h1 = x @ W1 ----
    cvt_f32_to_f16<<<nblk((long)N * F_IN, 256), 256, 0, stream>>>(x, x_h, N * F_IN);
    transpose_cvt_f16<<<nblk((long)F1 * F_IN, 256), 256, 0, stream>>>(W1, w1t, F_IN, F1);
    {
        long waves = (long)((N + 15) / 16) * (F1 / 64);
        wmma_gemm_f16<<<nblk(waves * 32, 256), 256, 0, stream>>>(x_h, w1t, h1, N, F_IN, F1);
    }

    // ---- layer 1 attention + aggregation ----
    node_scores<<<nblk((long)N * H1, 256), 256, 0, stream>>>(h1, a_src1, a_dst1, es1, ed1, N, H1, HID);
    fill_f32<<<nblk((long)N * H1, 256), 256, 0, stream>>>(m1, -INFINITY, N * H1);
    fill_f32<<<nblk((long)N * H1, 256), 256, 0, stream>>>(z1, 0.f, N * H1);
    fill_f32<<<nblk((long)N * F1, 256), 256, 0, stream>>>(agg1, 0.f, N * F1);
    edge_max   <<<nblk((long)Etot * H1, 256), 256, 0, stream>>>(srcp, dstp, E, Etot, es1, ed1, m1, H1);
    edge_expsum<<<nblk((long)Etot * H1, 256), 256, 0, stream>>>(srcp, dstp, E, Etot, es1, ed1, m1, z1, H1);
    edge_aggregate<<<Etot, 128, 0, stream>>>(srcp, dstp, E, Etot, es1, ed1, m1, z1, h1, agg1, H1, HID);
    bias_elu<<<nblk((long)N * F1, 256), 256, 0, stream>>>(agg1, b1, (float*)nullptr, h1a, N * F1, F1);

    // ---- layer 2 GEMM: h2 = elu(h1) @ W2 ----
    transpose_cvt_f16<<<nblk((long)HID * F1, 256), 256, 0, stream>>>(W2, w2t, F1, HID);
    {
        long waves = (long)((N + 15) / 16) * (HID / 64);
        wmma_gemm_f16<<<nblk(waves * 32, 256), 256, 0, stream>>>(h1a, w2t, h2, N, F1, HID);
    }

    // ---- layer 2 attention + aggregation (H=1) ----
    node_scores<<<nblk((long)N, 256), 256, 0, stream>>>(h2, a_src2, a_dst2, es2, ed2, N, 1, HID);
    fill_f32<<<nblk((long)N, 256), 256, 0, stream>>>(m2, -INFINITY, N);
    fill_f32<<<nblk((long)N, 256), 256, 0, stream>>>(z2, 0.f, N);
    fill_f32<<<nblk((long)N * HID, 256), 256, 0, stream>>>(agg2, 0.f, N * HID);
    edge_max   <<<nblk((long)Etot, 256), 256, 0, stream>>>(srcp, dstp, E, Etot, es2, ed2, m2, 1);
    edge_expsum<<<nblk((long)Etot, 256), 256, 0, stream>>>(srcp, dstp, E, Etot, es2, ed2, m2, z2, 1);
    edge_aggregate<<<Etot, 128, 0, stream>>>(srcp, dstp, E, Etot, es2, ed2, m2, z2, h2, agg2, 1, HID);
    bias_elu<<<nblk((long)N * HID, 256), 256, 0, stream>>>(agg2, b2, h2a, (_Float16*)nullptr, N * HID, HID);

    // ---- global mean pool + heads ----
    fill_f32<<<nblk((long)G * HID, 256), 256, 0, stream>>>(pool, 0.f, G * HID);
    fill_f32<<<1, 256, 0, stream>>>(cnt, 0.f, G);
    pool_accum<<<N, HID, 0, stream>>>(h2a, batch, pool, cnt, N, HID);
    heads<<<G, 64, 0, stream>>>(pool, cnt, Wp, bp, Wv, bv, (float*)d_out, G, HID, A);
}
